// GRUbiRNN_81097572483832
// MI455X (gfx1250) — matrix operand order repaired and checked by reading, not compiled, not run on hardware
//
#include <hip/hip_runtime.h>
#include <stdint.h>

// ---------------------------------------------------------------------------
// Bidirectional 2-layer GRU + LayerNorm, persistent-WGP design for MI455X.
//  - grid (16 batch-tiles x 2 directions), 256 threads (8 wave32) per WG
//  - all weights converted fp32->bf16 once into LDS (~234 KB of 320 KB WGP LDS)
//  - per-step matmuls on v_wmma_f32_16x16x32_bf16 (fp32 accum, fp32 state/LN)
//  - x tiles double-buffered into LDS one step ahead via
//    global_load_async_to_lds_b128 (ASYNCcnt / s_wait_asynccnt)
//  - wave w owns hidden columns [16w,16w+16): its r/z/n gate accumulators are
//    exactly the operands of its pointwise GRU update (no gate exchange)
//  - gate nonlinearities on v_exp_f32/v_rcp_f32/v_rsq_f32 (no IEEE div chains)
//  - all loop-invariant per-lane LDS reads (biases, LN gamma/beta) hoisted
// ---------------------------------------------------------------------------

typedef __bf16 bf16;
typedef __attribute__((ext_vector_type(16))) bf16  bf16x16;
typedef __attribute__((ext_vector_type(8)))  bf16  bf16x8;
typedef __attribute__((ext_vector_type(4)))  bf16  bf16x4;
typedef __attribute__((ext_vector_type(8)))  float f32x8;

namespace cfg {
constexpr int B = 256, T = 2048, D = 64, H0 = 128, H1 = 64;
constexpr int G0 = 3 * H0;   // 384 gate rows, layer 0
constexpr int G1 = 3 * H1;   // 192 gate rows, layer 1
constexpr int TB = 16;       // batch tile (M of WMMA)

// LDS row strides (elements), padded to 16B multiples for ds_load_b128 + bank spread
constexpr int S_WIH0 = 72, S_WHH0 = 136, S_WIH1 = 136, S_WHH1 = 72;
constexpr int S_XTB = 72, S_H0 = 132, S_H0B = 136, S_H1 = 68, S_H1B = 72;

constexpr size_t OFF_WIH0 = 0;
constexpr size_t OFF_WHH0 = OFF_WIH0 + (size_t)G0 * S_WIH0 * 2;
constexpr size_t OFF_WIH1 = OFF_WHH0 + (size_t)G0 * S_WHH0 * 2;
constexpr size_t OFF_WHH1 = OFF_WIH1 + (size_t)G1 * S_WIH1 * 2;
constexpr size_t OFF_BIH0 = OFF_WHH1 + (size_t)G1 * S_WHH1 * 2;
constexpr size_t OFF_BHH0 = OFF_BIH0 + (size_t)G0 * 4;
constexpr size_t OFF_BIH1 = OFF_BHH0 + (size_t)G0 * 4;
constexpr size_t OFF_BHH1 = OFF_BIH1 + (size_t)G1 * 4;
constexpr size_t OFF_LNG0 = OFF_BHH1 + (size_t)G1 * 4;
constexpr size_t OFF_LNB0 = OFF_LNG0 + (size_t)H0 * 4;
constexpr size_t OFF_LNG1 = OFF_LNB0 + (size_t)H0 * 4;
constexpr size_t OFF_LNB1 = OFF_LNG1 + (size_t)H1 * 4;
constexpr size_t OFF_XSTG = OFF_LNB1 + (size_t)H1 * 4;      // 2 x 4096 B raw fp32
constexpr size_t OFF_XTB  = OFF_XSTG + 2 * (size_t)TB * D * 4;
constexpr size_t OFF_H0   = OFF_XTB  + (size_t)TB * S_XTB * 2;
constexpr size_t OFF_H0B  = OFF_H0   + (size_t)TB * S_H0  * 4;
constexpr size_t OFF_H1   = OFF_H0B  + (size_t)TB * S_H0B * 2;
constexpr size_t OFF_H1B  = OFF_H1   + (size_t)TB * S_H1  * 4;
constexpr size_t OFF_PSUM = OFF_H1B  + (size_t)TB * S_H1B * 2;
constexpr size_t OFF_PSQ  = OFF_PSUM + (size_t)TB * 16 * 4;
constexpr size_t OFF_MU   = OFF_PSQ  + (size_t)TB * 16 * 4;
constexpr size_t OFF_RSTD = OFF_MU   + (size_t)TB * 4;
constexpr size_t SMEM_BYTES = OFF_RSTD + (size_t)TB * 4;    // ~271.4 KB < 320 KB
} // namespace cfg

__device__ __forceinline__ bf16x16 cat16(bf16x8 lo, bf16x8 hi) {
  return __builtin_shufflevector(lo, hi, 0,1,2,3,4,5,6,7,8,9,10,11,12,13,14,15);
}

// B fragment (KxN slice, K window [kbase,kbase+32)). W is LDS [N][stride] = Wᵀ of
// the K-major operand; lane holds column n, e=0..7 -> K=kbase+8*laneHi+e, e=8..15 -> K=kbase+16+8*laneHi+(e-8).
__device__ __forceinline__ bf16x16 load_b_frag(const bf16* W, int stride, int n,
                                               int kbase, int laneHi) {
  const bf16* p = W + n * stride + kbase + laneHi * 8;
  bf16x8 lo = *(const bf16x8*)(p);
  bf16x8 hi = *(const bf16x8*)(p + 16);
  return cat16(lo, hi);
}

// A fragment (16xK row-per-lane) from bf16 LDS [16][stride].
// e=0..7 -> K = kbase + 8*laneHi + e ; e=8..15 -> K = kbase + 16 + 8*laneHi + (e-8)
__device__ __forceinline__ bf16x16 load_a_frag_bf(const bf16* H, int stride, int m,
                                                  int kbase, int laneHi) {
  const bf16* p = H + m * stride + kbase + laneHi * 8;
  bf16x8 lo = *(const bf16x8*)(p);
  bf16x8 hi = *(const bf16x8*)(p + 16);
  return cat16(lo, hi);
}

__device__ __forceinline__ f32x8 wmma_bf16(bf16x16 a, bf16x16 b, f32x8 c) {
  return __builtin_amdgcn_wmma_f32_16x16x32_bf16(false, a, false, b, (short)0, c,
                                                 false, false);
}

// fast saturating nonlinearities: single v_exp_f32 / v_rcp_f32 each
__device__ __forceinline__ float sigmoidf_(float v) {
  return __builtin_amdgcn_rcpf(1.0f + __expf(-v));
}
__device__ __forceinline__ float tanhf_(float v) {
  // tanh(x) = 1 - 2/(1 + e^(2x));  exp overflow/underflow saturates correctly
  return 1.0f - 2.0f * __builtin_amdgcn_rcpf(1.0f + __expf(2.0f * v));
}

// Async global -> LDS copy (CDNA5 GLOBAL_LOAD_ASYNC_TO_LDS_B128, ASYNCcnt).
__device__ __forceinline__ void async_x_load(uint32_t lds_off, const float* gptr) {
  asm volatile("global_load_async_to_lds_b128 %0, %1, off"
               :: "v"(lds_off), "v"(gptr) : "memory");
}
__device__ __forceinline__ void wait_async0() {
  asm volatile("s_wait_asynccnt 0x0" ::: "memory");
}

__global__ __launch_bounds__(256, 1)
void gru_bidir_persistent(
    const float* __restrict__ x,
    const float* __restrict__ Wih_f0, const float* __restrict__ Whh_f0,
    const float* __restrict__ bih_f0, const float* __restrict__ bhh_f0,
    const float* __restrict__ Wih_b0, const float* __restrict__ Whh_b0,
    const float* __restrict__ bih_b0, const float* __restrict__ bhh_b0,
    const float* __restrict__ ln_g0,  const float* __restrict__ ln_b0,
    const float* __restrict__ Wih_f1, const float* __restrict__ Whh_f1,
    const float* __restrict__ bih_f1, const float* __restrict__ bhh_f1,
    const float* __restrict__ Wih_b1, const float* __restrict__ Whh_b1,
    const float* __restrict__ bih_b1, const float* __restrict__ bhh_b1,
    const float* __restrict__ ln_g1,  const float* __restrict__ ln_b1,
    float* __restrict__ hlast)       // [2][256][64]
{
  using namespace cfg;
  const int tid    = threadIdx.x;
  const int dir    = blockIdx.y;
  const int b0     = blockIdx.x * TB;
  const int w      = tid >> 5;
  const int lane   = tid & 31;
  const int laneHi = lane >> 4;
  const int lm     = lane & 15;

  const float* Wih0 = dir ? Wih_b0 : Wih_f0;
  const float* Whh0 = dir ? Whh_b0 : Whh_f0;
  const float* bih0 = dir ? bih_b0 : bih_f0;
  const float* bhh0 = dir ? bhh_b0 : bhh_f0;
  const float* Wih1 = dir ? Wih_b1 : Wih_f1;
  const float* Whh1 = dir ? Whh_b1 : Whh_f1;
  const float* bih1 = dir ? bih_b1 : bih_f1;
  const float* bhh1 = dir ? bhh_b1 : bhh_f1;

  extern __shared__ char smem[];
  bf16*  sWih0 = (bf16*)(smem + OFF_WIH0);
  bf16*  sWhh0 = (bf16*)(smem + OFF_WHH0);
  bf16*  sWih1 = (bf16*)(smem + OFF_WIH1);
  bf16*  sWhh1 = (bf16*)(smem + OFF_WHH1);
  float* sBih0 = (float*)(smem + OFF_BIH0);
  float* sBhh0 = (float*)(smem + OFF_BHH0);
  float* sBih1 = (float*)(smem + OFF_BIH1);
  float* sBhh1 = (float*)(smem + OFF_BHH1);
  float* sLng0 = (float*)(smem + OFF_LNG0);
  float* sLnb0 = (float*)(smem + OFF_LNB0);
  float* sLng1 = (float*)(smem + OFF_LNG1);
  float* sLnb1 = (float*)(smem + OFF_LNB1);
  float* sXstg = (float*)(smem + OFF_XSTG);     // [2][256][4] raw fp32 chunks
  bf16*  sXtB  = (bf16*)(smem + OFF_XTB);       // [16][72] bf16 x tile
  float* sH0   = (float*)(smem + OFF_H0);
  bf16*  sH0b  = (bf16*)(smem + OFF_H0B);
  float* sH1   = (float*)(smem + OFF_H1);
  bf16*  sH1b  = (bf16*)(smem + OFF_H1B);
  float* sPsum = (float*)(smem + OFF_PSUM);
  float* sPsq  = (float*)(smem + OFF_PSQ);
  float* sMu   = (float*)(smem + OFF_MU);
  float* sRstd = (float*)(smem + OFF_RSTD);

  // ---- one-time: convert this direction's weights fp32 -> bf16 into LDS ----
  for (int i = tid; i < G0 * D;  i += 256) sWih0[(i >> 6) * S_WIH0 + (i & 63)]  = (bf16)Wih0[i];
  for (int i = tid; i < G0 * H0; i += 256) sWhh0[(i >> 7) * S_WHH0 + (i & 127)] = (bf16)Whh0[i];
  for (int i = tid; i < G1 * H0; i += 256) sWih1[(i >> 7) * S_WIH1 + (i & 127)] = (bf16)Wih1[i];
  for (int i = tid; i < G1 * H1; i += 256) sWhh1[(i >> 6) * S_WHH1 + (i & 63)]  = (bf16)Whh1[i];
  for (int i = tid; i < G0; i += 256) { sBih0[i] = bih0[i]; sBhh0[i] = bhh0[i]; }
  for (int i = tid; i < G1; i += 256) { sBih1[i] = bih1[i]; sBhh1[i] = bhh1[i]; }
  for (int i = tid; i < H0; i += 256) { sLng0[i] = ln_g0[i]; sLnb0[i] = ln_b0[i]; }
  for (int i = tid; i < H1; i += 256) { sLng1[i] = ln_g1[i]; sLnb1[i] = ln_b1[i]; }
  for (int i = tid; i < TB * S_H0;  i += 256) sH0[i]  = 0.0f;
  for (int i = tid; i < TB * S_H0B; i += 256) sH0b[i] = (bf16)0.0f;
  for (int i = tid; i < TB * S_H1;  i += 256) sH1[i]  = 0.0f;
  for (int i = tid; i < TB * S_H1B; i += 256) sH1b[i] = (bf16)0.0f;

  // Async x staging: thread tid owns the 16-byte chunk (m = tid>>4, k = (tid&15)*4)
  const int mS  = tid >> 4;
  const int k4  = (tid & 15) * 4;
  const float* xrow = x + (size_t)(b0 + mS) * T * D + k4;
  const uint32_t stg = (uint32_t)(size_t)(smem + OFF_XSTG) + (uint32_t)tid * 16u;
  {
    const int t0 = dir ? (T - 1) : 0;
    async_x_load(stg, xrow + (size_t)t0 * D);     // tile for s=0 -> buffer 0
  }
  __syncthreads();

  // ---- hoist loop-invariant per-lane LDS reads into registers ----
  const int j0 = w * 16 + lm;              // layer-0 gate column owned by lane
  const float bR0  = sBih0[j0] + sBhh0[j0];
  const float bZ0  = sBih0[H0 + j0] + sBhh0[H0 + j0];
  const float biN0 = sBih0[2 * H0 + j0];
  const float bhN0 = sBhh0[2 * H0 + j0];
  const int j1 = (w & 3) * 16 + lm;        // layer-1 gate column (waves 0..3)
  const float bR1  = sBih1[j1] + sBhh1[j1];
  const float bZ1  = sBih1[H1 + j1] + sBhh1[H1 + j1];
  const float biN1 = sBih1[2 * H1 + j1];
  const float bhN1 = sBhh1[2 * H1 + j1];
  // normalize loops: thread's column is fixed (256 = 0 mod 128 / mod 64)
  const int jn0 = tid & (H0 - 1);
  const float g0 = sLng0[jn0], be0 = sLnb0[jn0];
  const int jn1 = tid & (H1 - 1);
  const float g1 = sLng1[jn1], be1 = sLnb1[jn1];

  const f32x8 fzero = {0.f, 0.f, 0.f, 0.f, 0.f, 0.f, 0.f, 0.f};

  for (int s = 0; s < T; ++s) {
    // drain our async (tile s); barrier makes the whole staged tile visible
    wait_async0();
    __syncthreads();

    // convert own raw chunk -> bf16 x tile, then kick off tile s+1 async copy
    {
      const float* sp = sXstg + (size_t)(s & 1) * (TB * D) + tid * 4;
      const float f0 = sp[0], f1 = sp[1], f2 = sp[2], f3 = sp[3];
      bf16x4 pk = {(bf16)f0, (bf16)f1, (bf16)f2, (bf16)f3};
      *(bf16x4*)(sXtB + mS * S_XTB + k4) = pk;
    }
    if (s + 1 < T) {
      const int tn = dir ? (T - 2 - s) : (s + 1);
      async_x_load(stg + (uint32_t)(((s + 1) & 1) * TB * D * 4),
                   xrow + (size_t)tn * D);
    }
    __syncthreads();

    // ================= layer 0: all 8 waves, wave w -> cols [16w,16w+16) ====
    {
      f32x8 giR = fzero, giZ = fzero, giN = fzero;
      f32x8 ghR = fzero, ghZ = fzero, ghN = fzero;
      const int nR = 0 * H0 + j0;
      const int nZ = 1 * H0 + j0;
      const int nN = 2 * H0 + j0;
#pragma unroll
      for (int kb = 0; kb < D; kb += 32) {
        bf16x16 a = load_a_frag_bf(sXtB, S_XTB, lm, kb, laneHi);
        giR = wmma_bf16(a, load_b_frag(sWih0, S_WIH0, nR, kb, laneHi), giR);
        giZ = wmma_bf16(a, load_b_frag(sWih0, S_WIH0, nZ, kb, laneHi), giZ);
        giN = wmma_bf16(a, load_b_frag(sWih0, S_WIH0, nN, kb, laneHi), giN);
      }
#pragma unroll
      for (int kb = 0; kb < H0; kb += 32) {
        bf16x16 a = load_a_frag_bf(sH0b, S_H0B, lm, kb, laneHi);
        ghR = wmma_bf16(a, load_b_frag(sWhh0, S_WHH0, nR, kb, laneHi), ghR);
        ghZ = wmma_bf16(a, load_b_frag(sWhh0, S_WHH0, nZ, kb, laneHi), ghZ);
        ghN = wmma_bf16(a, load_b_frag(sWhh0, S_WHH0, nN, kb, laneHi), ghN);
      }
      // pointwise GRU update: each (m,j0) owned by exactly one lane
#pragma unroll
      for (int v = 0; v < 8; ++v) {
        const int m = v + 8 * laneHi;
        const float r  = sigmoidf_(giR[v] + ghR[v] + bR0);
        const float zg = sigmoidf_(giZ[v] + ghZ[v] + bZ0);
        const float ng = tanhf_(giN[v] + biN0 + r * (ghN[v] + bhN0));
        const float hp = sH0[m * S_H0 + j0];
        sH0[m * S_H0 + j0] = (1.0f - zg) * ng + zg * hp;   // h_new (pre-LN)
      }
    }
    __syncthreads();

    // LayerNorm layer 0 (rows of 128): parallel sum/sumsq partials
    {
      const int m = tid >> 4, part = tid & 15;
      const float* rp = sH0 + m * S_H0 + part * (H0 / 16);
      float ss = 0.f, qq = 0.f;
#pragma unroll
      for (int jj = 0; jj < H0 / 16; ++jj) { const float v = rp[jj]; ss += v; qq += v * v; }
      sPsum[tid] = ss; sPsq[tid] = qq;
    }
    __syncthreads();
    if (tid < TB) {
      float ss = 0.f, qq = 0.f;
#pragma unroll
      for (int p = 0; p < 16; ++p) { ss += sPsum[tid * 16 + p]; qq += sPsq[tid * 16 + p]; }
      const float mu = ss * (1.0f / H0);
      sMu[tid]   = mu;
      sRstd[tid] = __builtin_amdgcn_rsqf(qq * (1.0f / H0) - mu * mu + 1e-5f);
    }
    __syncthreads();
#pragma unroll
    for (int k = 0; k < (TB * H0) / 256; ++k) {
      const int m = ((tid + k * 256) >> 7);
      const float val = (sH0[m * S_H0 + jn0] - sMu[m]) * sRstd[m] * g0 + be0;
      sH0[m * S_H0 + jn0]   = val;          // new h0 (= y0, layer-1 input)
      sH0b[m * S_H0B + jn0] = (bf16)val;
    }
    __syncthreads();

    // ================= layer 1: waves 0..3, wave w -> cols [16w,16w+16) =====
    if (w < 4) {
      f32x8 giR = fzero, giZ = fzero, giN = fzero;
      f32x8 ghR = fzero, ghZ = fzero, ghN = fzero;
      const int nR = 0 * H1 + j1;
      const int nZ = 1 * H1 + j1;
      const int nN = 2 * H1 + j1;
#pragma unroll
      for (int kb = 0; kb < H0; kb += 32) {
        bf16x16 a = load_a_frag_bf(sH0b, S_H0B, lm, kb, laneHi);
        giR = wmma_bf16(a, load_b_frag(sWih1, S_WIH1, nR, kb, laneHi), giR);
        giZ = wmma_bf16(a, load_b_frag(sWih1, S_WIH1, nZ, kb, laneHi), giZ);
        giN = wmma_bf16(a, load_b_frag(sWih1, S_WIH1, nN, kb, laneHi), giN);
      }
#pragma unroll
      for (int kb = 0; kb < H1; kb += 32) {
        bf16x16 a = load_a_frag_bf(sH1b, S_H1B, lm, kb, laneHi);
        ghR = wmma_bf16(a, load_b_frag(sWhh1, S_WHH1, nR, kb, laneHi), ghR);
        ghZ = wmma_bf16(a, load_b_frag(sWhh1, S_WHH1, nZ, kb, laneHi), ghZ);
        ghN = wmma_bf16(a, load_b_frag(sWhh1, S_WHH1, nN, kb, laneHi), ghN);
      }
#pragma unroll
      for (int v = 0; v < 8; ++v) {
        const int m = v + 8 * laneHi;
        const float r  = sigmoidf_(giR[v] + ghR[v] + bR1);
        const float zg = sigmoidf_(giZ[v] + ghZ[v] + bZ1);
        const float ng = tanhf_(giN[v] + biN1 + r * (ghN[v] + bhN1));
        const float hp = sH1[m * S_H1 + j1];
        sH1[m * S_H1 + j1] = (1.0f - zg) * ng + zg * hp;
      }
    }
    __syncthreads();

    // LayerNorm layer 1 (rows of 64): parallel sum/sumsq partials
    {
      const int m = tid >> 4, part = tid & 15;
      const float* rp = sH1 + m * S_H1 + part * (H1 / 16);
      float ss = 0.f, qq = 0.f;
#pragma unroll
      for (int jj = 0; jj < H1 / 16; ++jj) { const float v = rp[jj]; ss += v; qq += v * v; }
      sPsum[tid] = ss; sPsq[tid] = qq;
    }
    __syncthreads();
    if (tid < TB) {
      float ss = 0.f, qq = 0.f;
#pragma unroll
      for (int p = 0; p < 16; ++p) { ss += sPsum[tid * 16 + p]; qq += sPsq[tid * 16 + p]; }
      const float mu = ss * (1.0f / H1);
      sMu[tid]   = mu;
      sRstd[tid] = __builtin_amdgcn_rsqf(qq * (1.0f / H1) - mu * mu + 1e-5f);
    }
    __syncthreads();
#pragma unroll
    for (int k = 0; k < (TB * H1) / 256; ++k) {
      const int m = ((tid + k * 256) >> 6);
      const float val = (sH1[m * S_H1 + jn1] - sMu[m]) * sRstd[m] * g1 + be1;
      sH1[m * S_H1 + jn1]   = val;
      sH1b[m * S_H1B + jn1] = (bf16)val;
    }
    __syncthreads();
  }

  // final layer-1 hidden state -> workspace
  for (int i = tid; i < TB * H1; i += 256) {
    const int m = i >> 6, j = i & (H1 - 1);
    hlast[((size_t)dir * B + (b0 + m)) * H1 + j] = sH1[m * S_H1 + j];
  }
}

// ---------------------------------------------------------------------------
// head: out = relu(concat(hf,hb) @ fc1ᵀ + b1) @ fc2ᵀ + b2     (256 x 2)
// ---------------------------------------------------------------------------
__global__ __launch_bounds__(256, 1)
void fc_head(const float* __restrict__ hlast,
             const float* __restrict__ fc1W, const float* __restrict__ fc1b,
             const float* __restrict__ fc2W, const float* __restrict__ fc2b,
             float* __restrict__ out)
{
  __shared__ float w1[64 * 128];
  __shared__ float w2[2 * 64];
  __shared__ float b1[64];
  __shared__ float b2[2];
  const int tid = threadIdx.x;
  for (int i = tid; i < 64 * 128; i += 256) w1[i] = fc1W[i];
  for (int i = tid; i < 2 * 64;   i += 256) w2[i] = fc2W[i];
  if (tid < 64) b1[tid] = fc1b[tid];
  if (tid < 2)  b2[tid] = fc2b[tid];
  __syncthreads();

  const int b = tid;  // one batch row per thread, B == blockDim == 256
  float h[128];
#pragma unroll
  for (int j = 0; j < 64; ++j) h[j]      = hlast[(size_t)b * 64 + j];          // hf
#pragma unroll
  for (int j = 0; j < 64; ++j) h[64 + j] = hlast[(size_t)(256 + b) * 64 + j];  // hb

  float o0 = b2[0], o1 = b2[1];
  for (int o = 0; o < 64; ++o) {
    float a = b1[o];
#pragma unroll
    for (int k = 0; k < 128; ++k) a += h[k] * w1[o * 128 + k];
    a = fmaxf(a, 0.0f);
    o0 += a * w2[o];
    o1 += a * w2[64 + o];
  }
  out[b * 2 + 0] = o0;
  out[b * 2 + 1] = o1;
}

// ---------------------------------------------------------------------------
extern "C" void kernel_launch(void* const* d_in, const int* in_sizes, int n_in,
                              void* d_out, int out_size, void* d_ws, size_t ws_size,
                              hipStream_t stream) {
  (void)in_sizes; (void)n_in; (void)out_size; (void)ws_size;
  const float* x      = (const float*)d_in[0];
  const float* Wih_f0 = (const float*)d_in[1];
  const float* Whh_f0 = (const float*)d_in[2];
  const float* bih_f0 = (const float*)d_in[3];
  const float* bhh_f0 = (const float*)d_in[4];
  const float* Wih_b0 = (const float*)d_in[5];
  const float* Whh_b0 = (const float*)d_in[6];
  const float* bih_b0 = (const float*)d_in[7];
  const float* bhh_b0 = (const float*)d_in[8];
  const float* ln_g0  = (const float*)d_in[9];
  const float* ln_b0  = (const float*)d_in[10];
  const float* Wih_f1 = (const float*)d_in[11];
  const float* Whh_f1 = (const float*)d_in[12];
  const float* bih_f1 = (const float*)d_in[13];
  const float* bhh_f1 = (const float*)d_in[14];
  const float* Wih_b1 = (const float*)d_in[15];
  const float* Whh_b1 = (const float*)d_in[16];
  const float* bih_b1 = (const float*)d_in[17];
  const float* bhh_b1 = (const float*)d_in[18];
  const float* ln_g1  = (const float*)d_in[19];
  const float* ln_b1  = (const float*)d_in[20];
  const float* fc1W   = (const float*)d_in[21];
  const float* fc1b   = (const float*)d_in[22];
  const float* fc2W   = (const float*)d_in[23];
  const float* fc2b   = (const float*)d_in[24];

  float* hlast = (float*)d_ws;  // [2][256][64] fp32

  (void)hipFuncSetAttribute(reinterpret_cast<const void*>(gru_bidir_persistent),
                            hipFuncAttributeMaxDynamicSharedMemorySize,
                            (int)cfg::SMEM_BYTES);

  dim3 grid(cfg::B / cfg::TB, 2, 1);   // 16 batch tiles x 2 directions
  gru_bidir_persistent<<<grid, dim3(256), cfg::SMEM_BYTES, stream>>>(
      x, Wih_f0, Whh_f0, bih_f0, bhh_f0, Wih_b0, Whh_b0, bih_b0, bhh_b0,
      ln_g0, ln_b0, Wih_f1, Whh_f1, bih_f1, bhh_f1, Wih_b1, Whh_b1, bih_b1,
      bhh_b1, ln_g1, ln_b1, hlast);

  fc_head<<<dim3(1), dim3(256), 0, stream>>>(hlast, fc1W, fc1b, fc2W, fc2b,
                                             (float*)d_out);
}
